// _HierarchicalSoftmax_13632226197683
// MI455X (gfx1250) — matrix operand order repaired and checked by reading, not compiled, not run on hardware
//
#include <hip/hip_runtime.h>
#include <hip/hip_bf16.h>

// ---------------------------------------------------------------------------
// HierarchicalSoftmax on gfx1250 (MI455X), wave32 + v_wmma_f32_16x16x32_f16.
//
// Strategy: bucket rows by global occurrence class o = phase*8 + occ_local
// (128 classes). Each 16-row WMMA tile then needs exactly three 16-column
// weight blocks: phase head and p2o tile stream from L2 (f16, pre-converted,
// guaranteed L2-resident at 1.1 MB total), the per-class o2s tile stays
// resident in VGPRs. X is read once (gathered), logits never touch memory,
// softmax picks use one ds_bpermute broadcast from the selected lane.
// Invalid tail rows load a safe dummy row and are never written -- no exec
// masking in the hot loop. Wave stays under 256 VGPRs for occupancy.
// ---------------------------------------------------------------------------

#define NROWS 131072
#define HDIM  256
#define NCLS  128      // P*K_OCC global occurrence classes
#define GRIDY 32       // tile-stride grid for the main kernel

typedef __attribute__((ext_vector_type(16))) _Float16 v16h;
typedef __attribute__((ext_vector_type(8)))  float    v8f;

// ----- workspace layout (bytes) -----
// [0,8192)                 W_phase  f16   (16*256)
// [8192,73728)             W_p2o    f16   (128*256)
// [73728,1122304)          W_o2s    f16   (2048*256)
// [1122304, +512)          hist[128]
// [+512, +1024)            cursor[128]
// [+1024, +1536)           base[128]
// [+1536, +1536+4*NROWS)   perm[NROWS]

__device__ __forceinline__ float rmax16(float v) {
#pragma unroll
  for (int m = 8; m; m >>= 1) v = fmaxf(v, __shfl_xor(v, m, 32));
  return v;
}
__device__ __forceinline__ float rsum16(float v) {
#pragma unroll
  for (int m = 8; m; m >>= 1) v += __shfl_xor(v, m, 32);
  return v;
}

// --- kernel 1: fp32 -> f16 weight conversion (+ zero histogram) -------------
__global__ void k_convert(const float* __restrict__ Wph,
                          const float* __restrict__ Wp2o,
                          const float* __restrict__ Wo2s,
                          _Float16* __restrict__ dPh,
                          _Float16* __restrict__ dP2o,
                          _Float16* __restrict__ dO2s,
                          int* __restrict__ hist) {
  const int i = blockIdx.x * blockDim.x + threadIdx.x;
  if (i < NCLS) hist[i] = 0;
  if (i < 16 * HDIM) {
    dPh[i] = (_Float16)Wph[i];
  } else if (i < 16 * HDIM + 128 * HDIM) {
    const int j = i - 16 * HDIM;
    dP2o[j] = (_Float16)Wp2o[j];
  } else if (i < 16 * HDIM + 128 * HDIM + 2048 * HDIM) {
    const int j = i - (16 * HDIM + 128 * HDIM);
    dO2s[j] = (_Float16)Wo2s[j];
  }
}

// --- kernel 2: occurrence-class histogram -----------------------------------
__global__ void k_hist(const int* __restrict__ Xph, const int* __restrict__ Xoc,
                       int* __restrict__ hist) {
  __shared__ int h[NCLS];
  if (threadIdx.x < NCLS) h[threadIdx.x] = 0;
  __syncthreads();
  const int i = blockIdx.x * blockDim.x + threadIdx.x;
  if (i < NROWS) atomicAdd(&h[Xph[i] * 8 + Xoc[i]], 1);
  __syncthreads();
  if (threadIdx.x < NCLS) {
    int v = h[threadIdx.x];
    if (v) atomicAdd(&hist[threadIdx.x], v);
  }
}

// --- kernel 3: tiny serial exclusive scan (128 entries) ---------------------
__global__ void k_scan(const int* __restrict__ hist, int* __restrict__ basep,
                       int* __restrict__ cursor) {
  if (threadIdx.x == 0 && blockIdx.x == 0) {
    int run = 0;
#pragma unroll 1
    for (int i = 0; i < NCLS; ++i) {
      basep[i] = run;
      cursor[i] = run;
      run += hist[i];
    }
  }
}

// --- kernel 4: scatter row ids into class-sorted perm[] ---------------------
__global__ void k_scatter(const int* __restrict__ Xph, const int* __restrict__ Xoc,
                          int* __restrict__ cursor, int* __restrict__ perm) {
  const int i = blockIdx.x * blockDim.x + threadIdx.x;
  if (i < NROWS) {
    const int o = Xph[i] * 8 + Xoc[i];
    const int pos = atomicAdd(&cursor[o], 1);
    perm[pos] = i;
  }
}

// --- kernel 5: fused 3-head WMMA + softmax ----------------------------------
// grid = (128 classes, GRIDY), block = 32 (one wave). Each wave strides over
// 16-row tiles of its class; the o2s B fragments stay resident in VGPRs.
__global__ void __launch_bounds__(32)
k_main(const float* __restrict__ X,
       const float* __restrict__ bph_g, const float* __restrict__ bp2o_g,
       const float* __restrict__ bo2s_g,
       const int* __restrict__ Xsubj,
       const _Float16* __restrict__ WphH, const _Float16* __restrict__ Wp2oH,
       const _Float16* __restrict__ Wo2sH,
       const int* __restrict__ basep, const int* __restrict__ hist,
       const int* __restrict__ perm,
       float* __restrict__ out) {
  const int o   = blockIdx.x;            // occurrence class 0..127
  const int cnt = hist[o];
  if ((int)blockIdx.y * 16 >= cnt) return;
  const int first = basep[o];

  const int lane = threadIdx.x;          // 0..31 (wave32)
  const int n    = lane & 15;            // output column within 16-wide tile
  const int hi   = lane >> 4;            // half-wave id
  const int g    = hi << 4;              // shfl base of this lane's 16-group

  const int p      = o >> 3;             // phase of this class (uniform)
  const int occl   = o & 7;              // local occurrence (uniform)
  const int base8  = (p & 1) * 8;        // 8-block offset inside p2o tile
  const bool in8   = ((n >> 3) == (p & 1));
  const int selOcc = base8 + occl;

  // per-lane-column biases (uniform across tiles of this class)
  const float bP = bph_g[n];
  const float bO = bp2o_g[(o >> 4) * 16 + n];
  const float bS = bo2s_g[o * 16 + n];

  // resident o2s B fragments: B[k,n] = W[n,k]; lane holds 16 contiguous K.
  v16h Bsub[8];
  {
    const _Float16* wsb = Wo2sH + (size_t)(o * 16 + n) * HDIM + hi * 16;
#pragma unroll
    for (int c = 0; c < 8; ++c) Bsub[c] = *(const v16h*)(wsb + c * 32);
  }
  // streamed B pointers (L2-hot: whole converted weight set is 1.1 MB)
  const _Float16* wp = WphH  + (size_t)n * HDIM + hi * 16;                    // phase
  const _Float16* wo = Wp2oH + (size_t)((o >> 4) * 16 + n) * HDIM + hi * 16;  // p2o

#pragma unroll 1
  for (int tile = blockIdx.y; tile * 16 < cnt; tile += GRIDY) {
    const int tb = tile * 16;

    // lanes 0..15 own per-row metadata for this tile. Invalid tail rows map
    // to perm slot 0 (always exists: cnt>0 here) -> safe garbage, never stored.
    int rowL = 0, vldL = 0, sbjL = 0;
    if (lane < 16) {
      const int idx = tb + lane;
      vldL = (idx < cnt) ? 1 : 0;
      rowL = perm[first + (vldL ? idx : 0)];
      sbjL = Xsubj[rowL];
    }
    const int myRow = __shfl(rowL, n, 32);   // A-layout: lane holds row (lane&15)
    const float* xrow = X + (size_t)myRow * HDIM + hi * 16;

    v8f accP = {}; v8f accO = {}; v8f accS = {};
#pragma unroll
    for (int c = 0; c < 8; ++c) {            // K chunks of 32
      // A 16x32, contiguous k order (same as B): lane(hi) -> K = c*32 + hi*16 + {0..15}
      const float4* px = (const float4*)(xrow + c * 32);
      const float4 x0 = px[0], x1 = px[1], x2 = px[2], x3 = px[3];
      v16h a;
      a[0]=(_Float16)x0.x;  a[1]=(_Float16)x0.y;  a[2]=(_Float16)x0.z;  a[3]=(_Float16)x0.w;
      a[4]=(_Float16)x1.x;  a[5]=(_Float16)x1.y;  a[6]=(_Float16)x1.z;  a[7]=(_Float16)x1.w;
      a[8]=(_Float16)x2.x;  a[9]=(_Float16)x2.y;  a[10]=(_Float16)x2.z; a[11]=(_Float16)x2.w;
      a[12]=(_Float16)x3.x; a[13]=(_Float16)x3.y; a[14]=(_Float16)x3.z; a[15]=(_Float16)x3.w;

      const v16h bp = *(const v16h*)(wp + c * 32);
      const v16h bo = *(const v16h*)(wo + c * 32);
      accP = __builtin_amdgcn_wmma_f32_16x16x32_f16(false, a, false, bp,      (short)0, accP, false, false);
      accO = __builtin_amdgcn_wmma_f32_16x16x32_f16(false, a, false, bo,      (short)0, accO, false, false);
      accS = __builtin_amdgcn_wmma_f32_16x16x32_f16(false, a, false, Bsub[c], (short)0, accS, false, false);
    }

    // epilogue: C/D layout -> acc[r] holds row r (lanes 0-15) / r+8 (16-31)
#pragma unroll
    for (int r = 0; r < 8; ++r) {
      const int M   = r + hi * 8;
      const int vld = __shfl(vldL, M, 32);
      if (!vld) continue;                    // uniform within each half-wave
      const int sbj = __shfl(sbjL, M, 32);
      const int row = __shfl(rowL, M, 32);

      // phase head: softmax over 16, pick column p (uniform for this class)
      float l  = accP[r] + bP;
      float mx = rmax16(l);
      float e  = __expf(l - mx);
      float s  = rsum16(e);
      float pp = __shfl(e, g + p, 32) / s;

      // occurrence head: softmax over the 8-block [base8, base8+8)
      float lo = accO[r] + bO;
      float mo = rmax16(in8 ? lo : -1e30f);
      float eo = in8 ? __expf(lo - mo) : 0.0f;
      float so = rsum16(eo);
      float po = pp * (__shfl(eo, g + selOcc, 32) / so);

      // subject head: softmax over 16, pick per-row subj index
      float ls = accS[r] + bS;
      float ms = rmax16(ls);
      float es = __expf(ls - ms);
      float ss = rsum16(es);
      float ps = po * (__shfl(es, g + sbj, 32) / ss);

      if (n == 0) {
        out[row]             = pp;
        out[NROWS + row]     = po;
        out[2 * NROWS + row] = ps;
      }
    }
  }
}

extern "C" void kernel_launch(void* const* d_in, const int* in_sizes, int n_in,
                              void* d_out, int out_size, void* d_ws, size_t ws_size,
                              hipStream_t stream) {
  const float* X    = (const float*)d_in[0];
  const float* Wph  = (const float*)d_in[1];
  const float* bph  = (const float*)d_in[2];
  const float* Wp2o = (const float*)d_in[3];
  const float* bp2o = (const float*)d_in[4];
  const float* Wo2s = (const float*)d_in[5];
  const float* bo2s = (const float*)d_in[6];
  const int*   Xph  = (const int*)d_in[7];
  const int*   Xoc  = (const int*)d_in[8];
  const int*   Xsb  = (const int*)d_in[9];
  float* out = (float*)d_out;

  char* ws = (char*)d_ws;
  _Float16* WphH  = (_Float16*)(ws);
  _Float16* Wp2oH = (_Float16*)(ws + 8192);
  _Float16* Wo2sH = (_Float16*)(ws + 8192 + 65536);
  int* hist   = (int*)(ws + 8192 + 65536 + 1048576);
  int* cursor = hist + NCLS;
  int* basep  = cursor + NCLS;
  int* perm   = basep + NCLS;

  // total f16 elements to convert: (16+128+2048)*256 = 561152 -> 2192 blocks
  k_convert<<<2192, 256, 0, stream>>>(Wph, Wp2o, Wo2s, WphH, Wp2oH, Wo2sH, hist);
  k_hist   <<<NROWS / 256, 256, 0, stream>>>(Xph, Xoc, hist);
  k_scan   <<<1, 32, 0, stream>>>(hist, basep, cursor);
  k_scatter<<<NROWS / 256, 256, 0, stream>>>(Xph, Xoc, cursor, perm);

  dim3 grid(NCLS, GRIDY);
  k_main<<<grid, 32, 0, stream>>>(X, bph, bp2o, bo2s, Xsb,
                                  WphH, Wp2oH, Wo2sH,
                                  basep, hist, perm, out);
}
